// NeRFRenderer_70970039599823
// MI455X (gfx1250) — matrix operand order repaired and checked by reading, not compiled, not run on hardware
//
#include <hip/hip_runtime.h>
#include <math.h>

// Fully-fused NeRF renderer for gfx1250 (MI455X).
// - One wave (32 lanes) renders one ray: 128 samples as 8 tiles of M=16.
// - All matmuls use v_wmma_f32_16x16x32_f16 (f16 A/B, f32 accumulate).
// - Weight matrices live in registers as pre-built B fragments (weight-stationary).
// - Activations restage through per-wave LDS between layers (D layout -> A layout).
// - Transmittance cumprod via 16-wide shfl scan; final reductions via shfl_down.

typedef __attribute__((ext_vector_type(16))) _Float16 v16h;
typedef __attribute__((ext_vector_type(8)))  float    v8f;
typedef _Float16 h16;

#define WAVES_PER_BLOCK 8
#define LDS_STRIDE 3072   // per-wave LDS slice (bytes), 16B aligned

__device__ __forceinline__ float softplus_f(float x) {
  return (x > 20.f) ? x : log1pf(expf(x));
}
__device__ __forceinline__ float sigmoid_f(float x) {
  return 1.f / (1.f + expf(-x));
}
__device__ __forceinline__ v8f wmma_f16(v16h a, v16h b, v8f c) {
  // D = A(16x32 f16) * B(32x16 f16) + C(16x16 f32)
  return __builtin_amdgcn_wmma_f32_16x16x32_f16(
      /*neg_a=*/false, a, /*neg_b=*/false, b,
      /*c_mod=*/(short)0, c, /*reuse_a=*/false, /*reuse_b=*/false);
}

__global__ __launch_bounds__(256) void nerf_fused_kernel(
    const float* __restrict__ rays_o, const float* __restrict__ rays_d,
    const float* __restrict__ w1,  const float* __restrict__ b1,
    const float* __restrict__ w2,  const float* __restrict__ b2,
    const float* __restrict__ cw1, const float* __restrict__ cb1,
    const float* __restrict__ cw2, const float* __restrict__ cb2,
    float* __restrict__ out, int nrays)
{
  __shared__ __align__(16) unsigned char smem[WAVES_PER_BLOCK * LDS_STRIDE];

  const int lane = threadIdx.x & 31;
  const int wave = threadIdx.x >> 5;
  int ray = blockIdx.x * WAVES_PER_BLOCK + wave;
  if (ray >= nrays) ray = nrays - 1;   // duplicate work is idempotent

  // per-wave LDS slices
  h16*   Hl    = (h16*)  (smem + wave * LDS_STRIDE);          // 16x64 halfs (H1, reused for H2)
  h16*   featL = (h16*)  (smem + wave * LDS_STRIDE + 2048);   // 16x16 halfs
  float* densL = (float*)(smem + wave * LDS_STRIDE + 2560);   // 16 f32 (feat[...,0], exact)
  float* rgbL  = (float*)(smem + wave * LDS_STRIDE + 2624);   // 16x3 f32

  const int  nloc = lane & 15;
  const bool hi   = lane >= 16;

  // ---------------- ray setup: AABB near/far ----------------
  const float ox = rays_o[ray*3+0], oy = rays_o[ray*3+1], oz = rays_o[ray*3+2];
  const float dx = rays_d[ray*3+0], dy = rays_d[ray*3+1], dz = rays_d[ray*3+2];
  float nearv = -1e30f, farv = 1e30f;
  {
    float o3[3] = {ox, oy, oz}, d3[3] = {dx, dy, dz};
    #pragma unroll
    for (int a = 0; a < 3; ++a) {
      float inv = 1.f / (d3[a] + 1e-15f);
      float t1 = (-1.f - o3[a]) * inv;
      float t2 = ( 1.f - o3[a]) * inv;
      nearv = fmaxf(nearv, fminf(t1, t2));
      farv  = fminf(farv,  fmaxf(t1, t2));
    }
    nearv = fmaxf(nearv, 0.05f);
    farv  = fmaxf(farv, nearv + 1e-6f);
  }
  const float span = farv - nearv;

  // ---------------- weight-stationary B fragments ----------------
  // B(32x16) f16 layout: lanes 0-15 hold column n=lane, elem 0..7 <-> K=0..7, elem 8..15 <-> K=16..23;
  // lanes 16-31 hold column n=lane-16, elem 0..7 <-> K=8..15, elem 8..15 <-> K=24..31.
  v16h bw1[4];   // W1: 3x64, K padded to 32, four N-tiles
  #pragma unroll
  for (int t = 0; t < 4; ++t) {
    v16h f;
    #pragma unroll
    for (int e = 0; e < 16; ++e) f[e] = (h16)0.f;
    if (!hi) {                       // lanes 16-31 (K=8..15, 24..31) are all-zero (K pad)
      int n = t*16 + nloc;
      f[0] = (h16)w1[0*64 + n];
      f[1] = (h16)w1[1*64 + n];
      f[2] = (h16)w1[2*64 + n];
    }
    bw1[t] = f;
  }
  v16h bw2[2];   // W2: 64x16, two K-chunks of 32
  #pragma unroll
  for (int c = 0; c < 2; ++c) {
    v16h f;
    int k0 = 32*c + (hi ? 8 : 0);
    #pragma unroll
    for (int e = 0; e < 16; ++e) f[e] = (h16)w2[(k0+e+(e>=8?8:0))*16 + nloc];
    bw2[c] = f;
  }
  v16h bc1[4];   // CW1: 19x64, K padded to 32, four N-tiles
  #pragma unroll
  for (int t = 0; t < 4; ++t) {
    v16h f;
    int n = t*16 + nloc;
    int k0 = hi ? 8 : 0;
    #pragma unroll
    for (int e = 0; e < 16; ++e) {
      int k = k0 + e + (e >= 8 ? 8 : 0);
      f[e] = (k < 19) ? (h16)cw1[k*64 + n] : (h16)0.f;
    }
    bc1[t] = f;
  }
  v16h bc2[2];   // CW2: 64x3, N padded to 16, two K-chunks
  #pragma unroll
  for (int c = 0; c < 2; ++c) {
    v16h f;
    int k0 = 32*c + (hi ? 8 : 0);
    #pragma unroll
    for (int e = 0; e < 16; ++e)
      f[e] = (nloc < 3) ? (h16)cw2[(k0+e+(e>=8?8:0))*3 + nloc] : (h16)0.f;
    bc2[c] = f;
  }

  // ---------------- compositing state ----------------
  float T = 1.f;                          // running transmittance (wave-mirrored)
  float pr = 0.f, pg = 0.f, pb = 0.f;     // per-lane partial sums (lanes 0-15 used)
  float dep = 0.f, wsum = 0.f;

  // ---------------- march: 8 tiles of 16 samples ----------------
  for (int tile = 0; tile < 8; ++tile) {
    const int sbase = tile * 16;

    // A1: sample positions, 16x32 f16 A-fragment.
    // Lanes 0-15 hold row M=lane: elem 0..7 <-> K=0..7, elem 8..15 <-> K=16..23.
    // Lanes 16-31 hold K=8..15 / 24..31 -> all zero (K pad beyond 3).
    float zf = nearv + span * ((float)(sbase + nloc) * (1.f / 127.f));
    float px = fminf(fmaxf(ox + dx*zf, -1.f), 1.f);
    float py = fminf(fmaxf(oy + dy*zf, -1.f), 1.f);
    float pz = fminf(fmaxf(oz + dz*zf, -1.f), 1.f);
    v16h a1;
    #pragma unroll
    for (int e = 0; e < 16; ++e) a1[e] = (h16)0.f;
    if (!hi) { a1[0] = (h16)px; a1[1] = (h16)py; a1[2] = (h16)pz; }

    // density layer 1: H = relu(X*W1 + b1), store H (16x64) to LDS row-major
    #pragma unroll
    for (int t = 0; t < 4; ++t) {
      float bias = b1[t*16 + nloc];
      v8f c;
      #pragma unroll
      for (int e = 0; e < 8; ++e) c[e] = bias;
      v8f d = wmma_f16(a1, bw1[t], c);
      int n = t*16 + nloc;
      #pragma unroll
      for (int v = 0; v < 8; ++v) {
        int m = v + (hi ? 8 : 0);
        Hl[m*64 + n] = (h16)fmaxf(d[v], 0.f);
      }
    }
    __syncthreads();

    // density layer 2: feat = H*W2 + b2 (two K=32 chunks)
    {
      float bias = b2[nloc];
      v8f acc;
      #pragma unroll
      for (int e = 0; e < 8; ++e) acc[e] = bias;
      #pragma unroll
      for (int c = 0; c < 2; ++c) {
        v16h a2;
        const h16* row = Hl + nloc * 64;
        int kA = 32*c + (hi ? 8 : 0);   // A layout: e<8 -> K=kA+e ; e>=8 -> K=kA+16+(e-8)
        #pragma unroll
        for (int e = 0; e < 8; ++e) { a2[e] = row[kA + e]; a2[8+e] = row[kA + 16 + e]; }
        acc = wmma_f16(a2, bw2[c], acc);
      }
      #pragma unroll
      for (int v = 0; v < 8; ++v) {
        int m = v + (hi ? 8 : 0);
        featL[m*16 + nloc] = (h16)acc[v];
        if (nloc == 0) densL[m] = acc[v];   // exact f32 feat[...,0]
      }
    }
    __syncthreads();

    // color layer 1: in = concat(dir(3), feat(16)) -> K=19 padded to 32
    v16h ac;
    if (!hi) {
      int m = nloc;
      ac[0] = (h16)dx; ac[1] = (h16)dy; ac[2] = (h16)dz;          // K=0..2
      #pragma unroll
      for (int e = 3; e < 8; ++e) ac[e] = featL[m*16 + (e-3)];    // K=3..7  -> feat0..4
      ac[8]  = featL[m*16 + 13];                                  // K=16 -> feat13
      ac[9]  = featL[m*16 + 14];                                  // K=17 -> feat14
      ac[10] = featL[m*16 + 15];                                  // K=18 -> feat15
      #pragma unroll
      for (int e = 11; e < 16; ++e) ac[e] = (h16)0.f;             // K=19..23 pad
    } else {
      int m = nloc;
      #pragma unroll
      for (int e = 0; e < 8; ++e) ac[e] = featL[m*16 + 5 + e];    // K=8..15 -> feat5..12
      #pragma unroll
      for (int e = 8; e < 16; ++e) ac[e] = (h16)0.f;              // K=24..31 pad
    }
    #pragma unroll
    for (int t = 0; t < 4; ++t) {
      float bias = cb1[t*16 + nloc];
      v8f c;
      #pragma unroll
      for (int e = 0; e < 8; ++e) c[e] = bias;
      v8f d = wmma_f16(ac, bc1[t], c);
      int n = t*16 + nloc;
      #pragma unroll
      for (int v = 0; v < 8; ++v) {
        int m = v + (hi ? 8 : 0);
        Hl[m*64 + n] = (h16)fmaxf(d[v], 0.f);   // reuse Hl for H2
      }
    }
    __syncthreads();

    // color layer 2: rgb = sigmoid(H2*CW2 + cb2)
    {
      float bias = (nloc < 3) ? cb2[nloc] : 0.f;
      v8f acc;
      #pragma unroll
      for (int e = 0; e < 8; ++e) acc[e] = bias;
      #pragma unroll
      for (int c = 0; c < 2; ++c) {
        v16h a2;
        const h16* row = Hl + nloc * 64;
        int kA = 32*c + (hi ? 8 : 0);
        #pragma unroll
        for (int e = 0; e < 8; ++e) { a2[e] = row[kA + e]; a2[8+e] = row[kA + 16 + e]; }
        acc = wmma_f16(a2, bc2[c], acc);
      }
      if (nloc < 3) {
        #pragma unroll
        for (int v = 0; v < 8; ++v) {
          int m = v + (hi ? 8 : 0);
          rgbL[m*3 + nloc] = sigmoid_f(acc[v]);
        }
      }
    }
    __syncthreads();

    // composite this tile: lanes 0-15 own sample m=nloc (hi lanes mirror harmlessly)
    {
      int m = nloc;
      int s = sbase + m;
      float dens  = softplus_f(densL[m]);
      float delta = (s < 127) ? (span * (1.f / 127.f)) : (span * (1.f / 128.f));
      float alpha = 1.f - expf(-delta * dens);
      float q = 1.f - alpha + 1e-15f;
      // inclusive product scan over 16 lanes
      float p = q;
      #pragma unroll
      for (int off = 1; off < 16; off <<= 1) {
        float o = __shfl_up(p, off, 16);
        if (nloc >= off) p *= o;
      }
      float excl = __shfl_up(p, 1, 16);
      if (nloc == 0) excl = 1.f;
      float w = alpha * T * excl;
      float Tall = __shfl(p, 15, 16);
      if (w > 1e-4f) {                 // color mask (weights>1e-4)
        pr += w * rgbL[m*3 + 0];
        pg += w * rgbL[m*3 + 1];
        pb += w * rgbL[m*3 + 2];
      }
      dep  += w * ((float)s * (1.f / 127.f));   // ori_z == s/127 for uniform samples
      wsum += w;
      T *= Tall;
    }
    __syncthreads();
  }

  // reduce lanes 0-15 -> lane 0
  #pragma unroll
  for (int off = 8; off >= 1; off >>= 1) {
    pr   += __shfl_down(pr,   off, 16);
    pg   += __shfl_down(pg,   off, 16);
    pb   += __shfl_down(pb,   off, 16);
    dep  += __shfl_down(dep,  off, 16);
    wsum += __shfl_down(wsum, off, 16);
  }
  if (lane == 0) {
    float bg = 1.f - wsum;              // white background
    out[ray*3 + 0] = pr + bg;
    out[ray*3 + 1] = pg + bg;
    out[ray*3 + 2] = pb + bg;
    out[nrays*3 + ray] = dep;           // depth after pixels in flat output
  }
}

extern "C" void kernel_launch(void* const* d_in, const int* in_sizes, int n_in,
                              void* d_out, int out_size, void* d_ws, size_t ws_size,
                              hipStream_t stream) {
  const float* rays_o = (const float*)d_in[0];
  const float* rays_d = (const float*)d_in[1];
  const float* w1  = (const float*)d_in[2];
  const float* b1  = (const float*)d_in[3];
  const float* w2  = (const float*)d_in[4];
  const float* b2  = (const float*)d_in[5];
  const float* cw1 = (const float*)d_in[6];
  const float* cb1 = (const float*)d_in[7];
  const float* cw2 = (const float*)d_in[8];
  const float* cb2 = (const float*)d_in[9];

  int nrays = in_sizes[0] / 3;
  int blocks = (nrays + WAVES_PER_BLOCK - 1) / WAVES_PER_BLOCK;
  nerf_fused_kernel<<<blocks, 256, 0, stream>>>(
      rays_o, rays_d, w1, b1, w2, b2, cw1, cb1, cw2, cb2,
      (float*)d_out, nrays);
}
